// SpamClassifierSingleLstmCell_56736517980926
// MI455X (gfx1250) — matrix operand order repaired and checked by reading, not compiled, not run on hardware
//
#include <hip/hip_runtime.h>
#include <hip/hip_bf16.h>
#include <math.h>

// ---------------------------------------------------------------------------
// SpamClassifier single-LSTM-cell, B=1, 300 sequential steps.
// Persistent single-workgroup kernel (32 wave32s). Weights pre-swizzled to
// f16 CDNA5 WMMA A-fragment layout (L2-resident, ~3.25 MB). Per step:
// gates[2048] = [W_ih|W_hh] . [xt;h] via v_wmma_f32_16x16x32_f16, with every
// B column broadcasting the same 32-K chunk of [xt;h]; bias pre-loaded into
// the WMMA accumulator (C operand) in D-layout.
// ---------------------------------------------------------------------------

#define VOCAB 100000
#define EMB   300
#define HID   512
#define GATES (4 * HID)          // 2048
#define KIN   320                // EMB padded to multiple of 32
#define KTOT  (KIN + HID)        // 832
#define KC    (KTOT / 32)        // 26 K-chunks of 32
#define NTILE (GATES / 16)       // 128 output tiles of 16 gates
#define NSTEP EMB                // torch-faithful: EMB steps, not SEQ

typedef __attribute__((ext_vector_type(16))) _Float16 v16h;
typedef __attribute__((ext_vector_type(8)))  float    v8f;

// ---------------------------------------------------------------------------
// Prep kernel: swizzle [W_ih | 0pad | W_hh] (f32) into f16 A-fragments.
// A-fragment K order chosen here (not ISA table order; B matches) (16x32, 16 halves
// per lane): lanes 0-15 row=lane, elems 0..15 -> K 0..15
//            lanes 16-31 row=lane-16, elems 0..15 -> K 16..31
// Flat storage: Afrag[((tile*KC + kc)*32 + lane)*16 + e]
// ---------------------------------------------------------------------------
__global__ void prep_weights_kernel(const float* __restrict__ W_ih,
                                    const float* __restrict__ W_hh,
                                    _Float16* __restrict__ Afrag) {
    int gid = blockIdx.x * blockDim.x + threadIdx.x;      // one per (tile,kc,lane)
    if (gid >= NTILE * KC * 32) return;
    int lane = gid & 31;
    int kc   = (gid >> 5) % KC;
    int tile = gid / (32 * KC);
    int row  = tile * 16 + (lane & 15);

    _Float16 frag[16];
#pragma unroll
    for (int e = 0; e < 16; ++e) {
        int K  = e + ((lane >> 4) << 4);
        int kg = kc * 32 + K;
        float w;
        if (kg < KIN) w = (kg < EMB) ? W_ih[(size_t)row * EMB + kg] : 0.0f;
        else          w = W_hh[(size_t)row * HID + (kg - KIN)];
        frag[e] = (_Float16)w;
    }
    _Float16* dst = Afrag + (size_t)gid * 16;
#pragma unroll
    for (int e = 0; e < 16; ++e) dst[e] = frag[e];
}

__device__ __forceinline__ float sigmoidf_(float x) {
    return 1.0f / (1.0f + expf(-x));
}

// ---------------------------------------------------------------------------
// Main persistent kernel: 1 block, 1024 threads = 32 waves.
// ---------------------------------------------------------------------------
__global__ void __launch_bounds__(1024)
lstm_persistent_kernel(const int*   __restrict__ x,
                       const float* __restrict__ h0,
                       const float* __restrict__ c0,
                       const float* __restrict__ emb,
                       const float* __restrict__ b_ih,
                       const float* __restrict__ b_hh,
                       const float* __restrict__ fc_w,
                       const float* __restrict__ fc_b,
                       const _Float16* __restrict__ Afrag,
                       float* __restrict__ out) {
    __shared__ __align__(64) _Float16 xh[KTOT];   // [xt(300) | pad(20) | h(512)] f16
    __shared__ __align__(64) float gates[GATES];
    __shared__ __align__(64) float bias[GATES];
    __shared__ float cbuf[HID];
    __shared__ float hbuf[HID];
    __shared__ float red[HID];

    const int tid  = threadIdx.x;
    const int lane = tid & 31;
    const int wave = tid >> 5;

    // one-time init: fused bias, h/c state, f16 h, pad zone
    for (int j = tid; j < GATES; j += 1024) bias[j] = b_ih[j] + b_hh[j];
    if (tid < HID) {
        cbuf[tid] = c0[tid];
        float h = h0[tid];
        hbuf[tid] = h;
        xh[KIN + tid] = (_Float16)h;
    }
    if (tid < (KIN - EMB)) xh[EMB + tid] = (_Float16)0.0f;
    __syncthreads();

    for (int t = 0; t < NSTEP; ++t) {
        // xt = emb_table[x[0][t]]  (f32 -> f16 into LDS)
        const int tok = x[t];
        const float* erow = emb + (size_t)tok * EMB;
        for (int j = tid; j < EMB; j += 1024) xh[j] = (_Float16)erow[j];
        __syncthreads();

        // gates = [W_ih | W_hh] * [xt ; h] + bias  via WMMA
        for (int tile = wave; tile < NTILE; tile += 32) {
            // Accumulator pre-loaded with bias in D-layout:
            // VGPR v holds row v (lanes 0-15) / row v+8 (lanes 16-31);
            // broadcast LDS read, ignored columns get same (harmless) values.
            v8f acc = *(const v8f*)&bias[tile * 16 + ((lane >> 4) << 3)];
            const v16h* abase = (const v16h*)Afrag + (size_t)tile * KC * 32 + lane;
#pragma unroll 2
            for (int kc = 0; kc < KC; ++kc) {
                v16h a = abase[(size_t)kc * 32];
                if (kc + 4 < KC)   // warm L2/WGP$ ~4 chunks (4 KB) ahead
                    __builtin_prefetch((const void*)&abase[(size_t)(kc + 4) * 32], 0, 3);
                // every column of B = same 32-K chunk of [xt;h]:
                // lanes 0-15 read halves [kc*32 .. +15], lanes 16-31 [+16 .. +31]
                v16h b = *(const v16h*)(xh + kc * 32 + (lane & 16));
                acc = __builtin_amdgcn_wmma_f32_16x16x32_f16(
                        /*neg_a=*/false, a, /*neg_b=*/false, b,
                        /*c_mod=*/(short)0, acc,
                        /*reuse_a=*/false, /*reuse_b=*/false);
            }
            // Column N=0 of D lives in lanes 0 (rows 0-7) and 16 (rows 8-15).
            if ((lane & 15) == 0) {
                int mbase = tile * 16 + ((lane >> 4) << 3);
#pragma unroll
                for (int v = 0; v < 8; ++v)
                    gates[mbase + v] = acc[v];
            }
        }
        __syncthreads();

        // element-wise LSTM cell update (torch gate order i,f,g,o)
        if (tid < HID) {
            float gi = sigmoidf_(gates[tid]);
            float gf = sigmoidf_(gates[HID + tid]);
            float gg = tanhf(gates[2 * HID + tid]);
            float go = sigmoidf_(gates[3 * HID + tid]);
            float c  = gf * cbuf[tid] + gi * gg;
            float h  = go * tanhf(c);
            cbuf[tid] = c;
            hbuf[tid] = h;
            xh[KIN + tid] = (_Float16)h;
        }
        __syncthreads();
    }

    // out = sigmoid(h . fc_w + fc_b)
    if (tid < HID) red[tid] = hbuf[tid] * fc_w[tid];
    __syncthreads();
    for (int s = HID / 2; s > 0; s >>= 1) {
        if (tid < s) red[tid] += red[tid + s];
        __syncthreads();
    }
    if (tid == 0) out[0] = sigmoidf_(red[0] + fc_b[0]);
    if (tid < HID) {
        out[1 + tid]       = hbuf[tid];
        out[1 + HID + tid] = cbuf[tid];
    }
}

extern "C" void kernel_launch(void* const* d_in, const int* in_sizes, int n_in,
                              void* d_out, int out_size, void* d_ws, size_t ws_size,
                              hipStream_t stream) {
    (void)in_sizes; (void)n_in; (void)out_size; (void)ws_size;

    const int*   x    = (const int*)  d_in[0];
    const float* h0   = (const float*)d_in[1];
    const float* c0   = (const float*)d_in[2];
    const float* emb  = (const float*)d_in[3];
    const float* W_ih = (const float*)d_in[4];
    const float* W_hh = (const float*)d_in[5];
    const float* b_ih = (const float*)d_in[6];
    const float* b_hh = (const float*)d_in[7];
    const float* fc_w = (const float*)d_in[8];
    const float* fc_b = (const float*)d_in[9];
    float* out = (float*)d_out;

    _Float16* Afrag = (_Float16*)d_ws;   // NTILE*KC*512 halves = ~3.25 MB

    const int prepN = NTILE * KC * 32;
    prep_weights_kernel<<<(prepN + 255) / 256, 256, 0, stream>>>(W_ih, W_hh, Afrag);

    lstm_persistent_kernel<<<1, 1024, 0, stream>>>(
        x, h0, c0, emb, b_ih, b_hh, fc_w, fc_b, Afrag, out);
}
